// BasicMPNN_40613210750998
// MI455X (gfx1250) — matrix operand order repaired and checked by reading, not compiled, not run on hardware
//
#include <hip/hip_runtime.h>
#include <hip/hip_bf16.h>
#include <math.h>

// ---------------------------------------------------------------------------
// BasicMPNN on MI455X (gfx1250, wave32):
//   msg = relu(x @ Wm + bm)                      -> WMMA f16 GEMM (B in LDS)
//   agg = scatter_add(msg[src], dst)             -> f32 atomics (BW bound)
//   h   = relu(concat(x,agg) @ Wu + bu)          -> WMMA f16 GEMM, pooling fused
//   out = sigmoid(meanpool_per_graph(h) @ Wo+bo) -> tiny reduction
// Weights are staged into LDS per workgroup via GLOBAL_LOAD_ASYNC_TO_LDS
// (ASYNCcnt) when the toolchain exposes the builtin.
// ---------------------------------------------------------------------------

typedef __attribute__((ext_vector_type(16))) _Float16 v16h;
typedef __attribute__((ext_vector_type(8)))  _Float16 v8h;
typedef __attribute__((ext_vector_type(8)))  float    v8f;
typedef int v4i __attribute__((vector_size(16)));   // builtin's b128 payload type

#define N_NODES 50000
#define IN_DIM  128
#define HID_DIM 128
#define NGRAPHS 64
#define XA_STRIDE 256   // [x | agg] per node, f16

// ------------------------- async LDS staging helpers -----------------------

#if __has_builtin(__builtin_amdgcn_global_load_async_to_lds_b128)
#define HAVE_ASYNC_LDS 1
#else
#define HAVE_ASYNC_LDS 0
#endif

__device__ __forceinline__ void stage_b128(const _Float16* g, _Float16* l) {
#if HAVE_ASYNC_LDS
    __builtin_amdgcn_global_load_async_to_lds_b128(
        (__attribute__((address_space(1))) v4i*)g,
        (__attribute__((address_space(3))) v4i*)l, 0, 0);
#else
    *(v8h*)l = *(const v8h*)g;     // global_load_b128 + ds_store_b128
#endif
}

__device__ __forceinline__ void stage_wait() {
#if HAVE_ASYNC_LDS
#if __has_builtin(__builtin_amdgcn_s_wait_asynccnt)
    __builtin_amdgcn_s_wait_asynccnt(0);
#else
    asm volatile("s_wait_asynccnt 0" ::: "memory");
#endif
#endif
    __syncthreads();
}

// ------------------------------ utility kernels ----------------------------

__global__ void zero_f32(float* __restrict__ p, int n) {
    int t = blockIdx.x * blockDim.x + threadIdx.x;
    if (t < n) p[t] = 0.0f;
}

// Repack row-major W[k][n] (K = KC*32, H=128 cols) into per-lane v16h B
// fragments: flat index ((tile*KC + c)*32 + lane)*16 + i holds
// W[c*32 + (lane>>4)*8 + (i&7) + (i>>3)*16][tile*16 + (lane&15)]  (ISA B-matrix layout).
__global__ void pack_w_kernel(const float* __restrict__ W, _Float16* __restrict__ dst,
                              int KC, int H) {
    int t = blockIdx.x * blockDim.x + threadIdx.x;
    int total = (H / 16) * KC * 32 * 16;
    if (t >= total) return;
    int i    = t & 15;
    int L    = (t >> 4) & 31;
    int rest = t >> 9;
    int c    = rest % KC;
    int tile = rest / KC;
    int k = c * 32 + (L >> 4) * 8 + (i & 7) + (i >> 3) * 16;
    int n = tile * 16 + (L & 15);
    dst[t] = (_Float16)W[k * H + n];
}

// xa[:, 0:128] = f16(x)
__global__ void convert_x_kernel(const float* __restrict__ x, _Float16* __restrict__ xa) {
    int t = blockIdx.x * blockDim.x + threadIdx.x;
    if (t >= N_NODES * IN_DIM) return;
    int n = t >> 7, c = t & 127;
    xa[n * XA_STRIDE + c] = (_Float16)x[t];
}

// xa[:, 128:256] = f16(agg)
__global__ void convert_agg_kernel(const float* __restrict__ agg, _Float16* __restrict__ xa) {
    int t = blockIdx.x * blockDim.x + threadIdx.x;
    if (t >= N_NODES * HID_DIM) return;
    int n = t >> 7, c = t & 127;
    xa[n * XA_STRIDE + 128 + c] = (_Float16)agg[t];
}

__global__ void count_kernel(const int* __restrict__ batch, float* __restrict__ counts) {
    int n = blockIdx.x * blockDim.x + threadIdx.x;
    if (n < N_NODES) atomicAdd(&counts[batch[n]], 1.0f);
}

// ------------------------- GEMM1: msg = relu(x@Wm+bm) ----------------------
// 8 waves/block share the 32KB packed-Wm image in LDS (async-staged).
// One wave -> 16 rows x 128 cols: 4 K-chunks x 8 N-tiles = 32 WMMAs.

__global__ void __launch_bounds__(256)
gemm_msg_kernel(const _Float16* __restrict__ xa,
                const _Float16* __restrict__ WP,
                const float* __restrict__ bias,
                float* __restrict__ msg) {
    __shared__ v16h ldsB[8 * 4 * 32];                  // 32 KB
    {   // cooperative stage: 2048 x b128, 8 per thread
        _Float16* l = (_Float16*)ldsB;
        int t = threadIdx.x;
#pragma unroll
        for (int i = 0; i < 8; ++i)
            stage_b128(WP + (size_t)(t + i * 256) * 8, l + (size_t)(t + i * 256) * 8);
        stage_wait();
    }
    int wave = (blockIdx.x * blockDim.x + threadIdx.x) >> 5;
    int lane = threadIdx.x & 31;
    if (wave < N_NODES / 16) {                          // wave-uniform guard
        int m0  = wave * 16;
        int nlo = lane & 15;
        int hi  = lane >> 4;
        const _Float16* arow = xa + (size_t)(m0 + nlo) * XA_STRIDE;

        v16h a[4];
#pragma unroll
        for (int c = 0; c < 4; ++c) {
            int k0 = c * 32 + hi * 8;
            v8h a0 = *(const v8h*)(arow + k0);
            v8h a1 = *(const v8h*)(arow + k0 + 16);
            a[c] = __builtin_shufflevector(a0, a1, 0,1,2,3,4,5,6,7,8,9,10,11,12,13,14,15);
        }
        v8f acc[8] = {};
#pragma unroll
        for (int c = 0; c < 4; ++c)
#pragma unroll
            for (int t = 0; t < 8; ++t) {
                v16h b = ldsB[(t * 4 + c) * 32 + lane];
                acc[t] = __builtin_amdgcn_wmma_f32_16x16x32_f16(
                    false, a[c], false, b, (short)0, acc[t], false, false);
            }
#pragma unroll
        for (int t = 0; t < 8; ++t) {
            int col = t * 16 + nlo;
            float bz = bias[col];
#pragma unroll
            for (int r = 0; r < 8; ++r) {
                int row = m0 + r + hi * 8;
                float v = acc[t][r] + bz;
                msg[(size_t)row * HID_DIM + col] = v > 0.f ? v : 0.f;
            }
        }
    }
}

// -------------------- scatter: agg[dst] += msg[src] ------------------------
// One wave per edge; each lane carries 4 channels (float4 gather + 4 atomics).

__global__ void scatter_kernel(const int* __restrict__ edge_index,
                               const float* __restrict__ msg,
                               float* __restrict__ agg, int n_edges) {
    int tid  = blockIdx.x * blockDim.x + threadIdx.x;
    int e    = tid >> 5;
    int lane = tid & 31;
    if (e >= n_edges) return;
    int dst = edge_index[e];              // edge_index[0][e]
    int src = edge_index[n_edges + e];    // edge_index[1][e]
    int c0  = lane * 4;
    const float4 m4 = *(const float4*)(msg + (size_t)src * HID_DIM + c0);
    float* a = agg + (size_t)dst * HID_DIM + c0;
    atomicAdd(a + 0, m4.x);
    atomicAdd(a + 1, m4.y);
    atomicAdd(a + 2, m4.z);
    atomicAdd(a + 3, m4.w);
}

// ---------------- GEMM2: h = relu(concat(x,agg)@Wu+bu), fused pooling ------
// 64KB packed Wu staged in LDS; K=256 (8 chunks) x 8 N-tiles = 64 WMMAs/wave.
// Epilogue adds relu(h) rows straight into per-graph pooled sums.

__global__ void __launch_bounds__(256)
gemm_update_kernel(const _Float16* __restrict__ xa,
                   const _Float16* __restrict__ WP,
                   const float* __restrict__ bias,
                   const int* __restrict__ batch,
                   float* __restrict__ pooled) {
    __shared__ v16h ldsB[8 * 8 * 32];                  // 64 KB
    {   // cooperative stage: 4096 x b128, 16 per thread
        _Float16* l = (_Float16*)ldsB;
        int t = threadIdx.x;
#pragma unroll
        for (int i = 0; i < 16; ++i)
            stage_b128(WP + (size_t)(t + i * 256) * 8, l + (size_t)(t + i * 256) * 8);
        stage_wait();
    }
    int wave = (blockIdx.x * blockDim.x + threadIdx.x) >> 5;
    int lane = threadIdx.x & 31;
    if (wave < N_NODES / 16) {
        int m0  = wave * 16;
        int nlo = lane & 15;
        int hi  = lane >> 4;
        const _Float16* arow = xa + (size_t)(m0 + nlo) * XA_STRIDE;

        v16h a[8];
#pragma unroll
        for (int c = 0; c < 8; ++c) {
            int k0 = c * 32 + hi * 8;
            v8h a0 = *(const v8h*)(arow + k0);
            v8h a1 = *(const v8h*)(arow + k0 + 16);
            a[c] = __builtin_shufflevector(a0, a1, 0,1,2,3,4,5,6,7,8,9,10,11,12,13,14,15);
        }
        v8f acc[8] = {};
#pragma unroll
        for (int c = 0; c < 8; ++c)
#pragma unroll
            for (int t = 0; t < 8; ++t) {
                v16h b = ldsB[(t * 8 + c) * 32 + lane];
                acc[t] = __builtin_amdgcn_wmma_f32_16x16x32_f16(
                    false, a[c], false, b, (short)0, acc[t], false, false);
            }
        int gb[8];
#pragma unroll
        for (int r = 0; r < 8; ++r) gb[r] = batch[m0 + r + hi * 8];
#pragma unroll
        for (int t = 0; t < 8; ++t) {
            int col = t * 16 + nlo;
            float bz = bias[col];
#pragma unroll
            for (int r = 0; r < 8; ++r) {
                float v = acc[t][r] + bz;
                v = v > 0.f ? v : 0.f;
                atomicAdd(&pooled[gb[r] * HID_DIM + col], v);
            }
        }
    }
}

// --------------------- head: sigmoid(meanpool @ Wo + bo) -------------------

__global__ void final_kernel(const float* __restrict__ pooled,
                             const float* __restrict__ counts,
                             const float* __restrict__ Wo,
                             const float* __restrict__ bo,
                             float* __restrict__ out) {
    __shared__ float red[128];
    int g = blockIdx.x, c = threadIdx.x;
    float cnt = counts[g];
    cnt = cnt > 1.f ? cnt : 1.f;
    red[c] = (pooled[g * HID_DIM + c] / cnt) * Wo[c];
    __syncthreads();
    for (int s = 64; s > 0; s >>= 1) {
        if (c < s) red[c] += red[c + s];
        __syncthreads();
    }
    if (c == 0) out[g] = 1.f / (1.f + expf(-(red[0] + bo[0])));
}

// ------------------------------- launcher ----------------------------------

extern "C" void kernel_launch(void* const* d_in, const int* in_sizes, int n_in,
                              void* d_out, int out_size, void* d_ws, size_t ws_size,
                              hipStream_t stream) {
    const float* x     = (const float*)d_in[0];
    const int*   eidx  = (const int*)  d_in[1];
    const int*   batch = (const int*)  d_in[2];
    const float* Wm    = (const float*)d_in[3];
    const float* bm    = (const float*)d_in[4];
    const float* Wu    = (const float*)d_in[5];
    const float* bu    = (const float*)d_in[6];
    const float* Wo    = (const float*)d_in[7];
    const float* bo    = (const float*)d_in[8];
    float* out = (float*)d_out;
    const int n_edges = in_sizes[1] / 2;

    // workspace carve-out (256B aligned slabs)
    char* ws = (char*)d_ws;
    size_t off = 0;
    auto carve = [&](size_t bytes) { size_t r = off; off += (bytes + 255) & ~(size_t)255; return r; };
    _Float16* xa     = (_Float16*)(ws + carve((size_t)N_NODES * XA_STRIDE * 2)); // 25.6 MB
    float*    msg    = (float*)   (ws + carve((size_t)N_NODES * HID_DIM * 4));   // 25.6 MB
    float*    agg    = (float*)   (ws + carve((size_t)N_NODES * HID_DIM * 4));   // 25.6 MB
    _Float16* WmP    = (_Float16*)(ws + carve(128 * 128 * 2));                   // 32 KB
    _Float16* WuP    = (_Float16*)(ws + carve(256 * 128 * 2));                   // 64 KB
    float*    pooled = (float*)   (ws + carve(NGRAPHS * HID_DIM * 4));
    float*    counts = (float*)   (ws + carve(NGRAPHS * 4));

    const int B = 256;
    // 1) zero accumulators (atomics accumulate, ws is not re-poisoned)
    zero_f32<<<(N_NODES * HID_DIM + B - 1) / B, B, 0, stream>>>(agg, N_NODES * HID_DIM);
    zero_f32<<<(NGRAPHS * HID_DIM + B - 1) / B, B, 0, stream>>>(pooled, NGRAPHS * HID_DIM);
    zero_f32<<<1, 64, 0, stream>>>(counts, NGRAPHS);
    // 2) repack weights into WMMA B-fragment layout (f16)
    pack_w_kernel<<<(8 * 4 * 512 + B - 1) / B, B, 0, stream>>>(Wm, WmP, 4, 128);
    pack_w_kernel<<<(8 * 8 * 512 + B - 1) / B, B, 0, stream>>>(Wu, WuP, 8, 128);
    // 3) activations -> f16 (x half of the concat buffer) + per-graph counts
    convert_x_kernel<<<(N_NODES * IN_DIM + B - 1) / B, B, 0, stream>>>(x, xa);
    count_kernel<<<(N_NODES + B - 1) / B, B, 0, stream>>>(batch, counts);
    // 4) GEMM1 (WMMA, LDS-staged weights): msg = relu(x@Wm+bm)
    int waves = N_NODES / 16;                       // 3125, exact
    gemm_msg_kernel<<<(waves * 32 + B - 1) / B, B, 0, stream>>>(xa, WmP, bm, msg);
    // 5) edge scatter-add (memory-bound phase)
    scatter_kernel<<<((size_t)n_edges * 32 + B - 1) / B, B, 0, stream>>>(eidx, msg, agg, n_edges);
    // 6) agg -> f16 into concat buffer
    convert_agg_kernel<<<(N_NODES * HID_DIM + B - 1) / B, B, 0, stream>>>(agg, xa);
    // 7) GEMM2 (WMMA, LDS-staged weights) with fused mean-pool accumulation
    gemm_update_kernel<<<(waves * 32 + B - 1) / B, B, 0, stream>>>(xa, WuP, bu, batch, pooled);
    // 8) head
    final_kernel<<<NGRAPHS, 128, 0, stream>>>(pooled, counts, Wo, bo, out);
}
